// Net_31044023616363
// MI455X (gfx1250) — matrix operand order repaired and checked by reading, not compiled, not run on hardware
//
#include <hip/hip_runtime.h>
#include <hip/hip_bf16.h>

// ---------------------------------------------------------------------------
// MI455X (gfx1250) 4-frame deformable fusion network.
// All 3x3 convs -> implicit GEMM on v_wmma_f32_16x16x32_f16:
//   - A fragments staged through LDS (channel-major) -> 2x ds_load_b128 each,
//     hoisted into registers by the compiler and reused across all n-tiles
//   - B fragments pre-packed f16 in wave-fragment layout -> clause'd b128
//     loads; for NK<=9 the whole B tile is preloaded into a live register
//     array so waits stagger (s_wait_loadcnt N) instead of full drains
//   - K padded so each 32-chunk sits inside one 3x3 tap (no runtime decode)
// DCN3D sampling + coord-attention stay fp32 VALU (gather-bound / tiny).
// H = W = 256, B = 1.
// ---------------------------------------------------------------------------

typedef __attribute__((ext_vector_type(16))) _Float16 v16h;
typedef __attribute__((ext_vector_type(8)))  _Float16 v8h;
typedef __attribute__((ext_vector_type(8)))  float    v8f;
typedef __attribute__((ext_vector_type(4)))  float    v4f;

#define HW    65536
#define HDIM  256
#define WDIM  256

__device__ __forceinline__ float sigmoidf_(float v) { return 1.0f / (1.0f + expf(-v)); }
__device__ __forceinline__ float hswishf_(float v) {
    float r = fminf(fmaxf(v + 3.0f, 0.0f), 6.0f);
    return v * r * (1.0f / 6.0f);
}

// ---------------------------------------------------------------------------
// Weight pre-pack: f32 [Cout][CIN][3][3] -> f16 fragments, layout
//   bpack[((ntile*NK + kc)*32 + lane)*16 + j]
// B-fragment element j at lane: k = kc*32 + (j<8 ? (lane>>4)*8 + j : 16 + (lane>>4)*8 + j-8) ; zero-padded for
// ci >= CIN (channel pad to CINP) and n >= Cout.
// ---------------------------------------------------------------------------
template<int CINP, int CIN>
__global__ void pack_w_k(const float* __restrict__ w, _Float16* __restrict__ bpack,
                         int Cout)
{
    constexpr int NK  = (CINP * 9) / 32;
    constexpr int CPT = CINP / 32;              // k-chunks per 3x3 tap
    int tid = blockIdx.x * blockDim.x + threadIdx.x;
    int ntiles = (Cout + 15) >> 4;
    if (tid >= ntiles * NK * 512) return;
    int j    = tid & 15;
    int lane = (tid >> 4) & 31;
    int kcg  = tid >> 9;                        // ntile*NK + kc
    int kc   = kcg % NK;
    int nt   = kcg / NK;
    int half = lane >> 4, lm = lane & 15;
    int n    = (nt << 4) + lm;
    int tap  = kc / CPT;
    int ci   = (kc % CPT) * 32 + (j < 8 ? half * 8 + j : 16 + half * 8 + (j - 8));
    float v = 0.0f;
    if (ci < CIN && n < Cout) v = w[(n * CIN + ci) * 9 + tap];
    bpack[tid] = (_Float16)v;
}

// ---------------------------------------------------------------------------
// 3x3 SAME conv, implicit GEMM via WMMA.
// Block = one 16-pixel m-tile (one row segment), 4 waves share the LDS-staged
// input halo [3 rows][18 cols][CINP ch] (f16, channel-major => contiguous A).
// Wave w handles n-tiles w, w+4, ... ; kc fully unrolled.
// ---------------------------------------------------------------------------
template<int CINP, int CIN, int ACT, bool HAS_SKIP>
__global__ __launch_bounds__(128)
void conv3x3_wmma(const float* __restrict__ in, const _Float16* __restrict__ bpack,
                  const float* __restrict__ bias, const float* __restrict__ skip,
                  float* __restrict__ out, int Cout)
{
    constexpr int NK   = (CINP * 9) / 32;
    constexpr int CPT  = CINP / 32;
    constexpr int LOGC = (CINP == 64) ? 6 : 5;

    __shared__ __align__(16) _Float16 smem[54 * CINP];   // [yy][xx][ci]

    const int mtile = blockIdx.x;                // 4096 tiles of 16 pixels
    const int y0 = mtile >> 4;                   // tile row
    const int x0 = (mtile & 15) << 4;            // tile col start

    // ---- cooperative halo staging (bounds + channel pad resolved here) ----
    for (int idx = threadIdx.x; idx < 54 * CINP; idx += 128) {
        int ci  = idx & (CINP - 1);
        int pos = idx >> LOGC;                   // 0..53
        int yy  = pos / 18;
        int xx  = pos - yy * 18;
        int gy  = y0 + yy - 1;
        int gx  = x0 + xx - 1;
        float v = 0.0f;
        if (ci < CIN && (unsigned)gy < 256u && (unsigned)gx < 256u)
            v = in[ci * HW + gy * WDIM + gx];
        smem[idx] = (_Float16)v;
    }
    __syncthreads();

    const int lane = threadIdx.x & 31;
    const int wv   = threadIdx.x >> 5;
    const int half = lane >> 4;
    const int lm   = lane & 15;
    const int ntiles = (Cout + 15) >> 4;

    // per-lane A base: pixel column lm, channel sub-base half*8
    const _Float16* abase = smem + lm * CINP + half * 8;

    // A fragment for chunk kc (LDS, compile-time offsets; loop-invariant in nt)
    auto afrag = [&](int kc) -> v16h {
        const int tap    = kc / CPT;
        const int cibase = (kc % CPT) * 32;
        const int ky = tap / 3, kx = tap % 3;
        const int aoff = (ky * 18 + kx) * CINP + cibase;
        v8h lo = *(const v8h*)(abase + aoff);            // ci .. ci+7
        v8h hi = *(const v8h*)(abase + aoff + 16);       // ci+16 .. ci+23
        return __builtin_shufflevector(lo, hi,
                   0, 1, 2, 3, 4, 5, 6, 7, 8, 9, 10, 11, 12, 13, 14, 15);
    };

    for (int nt = wv; nt < ntiles; nt += 4) {    // wave-uniform loop
        const int nB = (nt << 4) + lm;
        v8f acc = {0.f, 0.f, 0.f, 0.f, 0.f, 0.f, 0.f, 0.f};
        const _Float16* bptr = bpack + ((nt * NK) * 32 + lane) * 16;

        if constexpr (NK <= 9) {
            // Preload the full B tile: all fragments live at once -> distinct
            // registers, loads issue ahead, staggered s_wait_loadcnt.
            v16h bfr[NK];
#pragma unroll
            for (int kc = 0; kc < NK; ++kc)
                bfr[kc] = *(const v16h*)(bptr + kc * 512);
#pragma unroll
            for (int kc = 0; kc < NK; ++kc)
                acc = __builtin_amdgcn_wmma_f32_16x16x32_f16(
                          false, afrag(kc), false, bfr[kc], (short)0, acc,
                          false, false);
        } else {
            // NK=18: full preload would spill; stream B per chunk.
            v16h bcur = *(const v16h*)(bptr);
#pragma unroll
            for (int kc = 0; kc < NK; ++kc) {
                v16h bnext = bcur;
                if (kc + 1 < NK) bnext = *(const v16h*)(bptr + (kc + 1) * 512);
                acc = __builtin_amdgcn_wmma_f32_16x16x32_f16(
                          false, afrag(kc), false, bcur, (short)0, acc,
                          false, false);
                bcur = bnext;
            }
        }

        if (nB < Cout) {
            const float bb = bias[nB];
            float* op = out + nB * HW + (mtile << 4) + (half << 3);   // 32B aligned
            float res[8];
            if (HAS_SKIP) {
                v4f s0 = *(const v4f*)(skip + nB * HW + (mtile << 4) + (half << 3));
                v4f s1 = *(const v4f*)(skip + nB * HW + (mtile << 4) + (half << 3) + 4);
#pragma unroll
                for (int r = 0; r < 8; ++r)
                    res[r] = acc[r] + bb + (r < 4 ? s0[r] : s1[r - 4]);
            } else {
#pragma unroll
                for (int r = 0; r < 8; ++r) res[r] = acc[r] + bb;
            }
            if (ACT == 1) {
#pragma unroll
                for (int r = 0; r < 8; ++r) res[r] = fmaxf(res[r], 0.0f);
            }
            v4f o0 = {res[0], res[1], res[2], res[3]};
            v4f o1 = {res[4], res[5], res[6], res[7]};
            *(v4f*)(op)     = o0;                // 2x global_store_b128
            *(v4f*)(op + 4) = o1;
        }
    }
}

template<int CINP, int CIN, int ACT, bool SK>
static void conv3x3(const float* in, const float* w, const float* b, const float* skip,
                    float* out, int Cout, _Float16* bpack, hipStream_t s)
{
    constexpr int NK = (CINP * 9) / 32;
    int ntiles = (Cout + 15) / 16;
    int total  = ntiles * NK * 512;
    pack_w_k<CINP, CIN><<<(total + 255) / 256, 256, 0, s>>>(w, bpack, Cout);
    conv3x3_wmma<CINP, CIN, ACT, SK><<<4096, 128, 0, s>>>(in, bpack, b, skip, out, Cout);
}

// ---------------------------------------------------------------------------
// DCN3D: trilinear deformable sampling over D=4 frames, 5x5 grid (N=100).
//   frames: [4][C][HW], raw: [400][HW] (off d/y/x = ch 0..299, mask 300..399)
//   w: [O][C][100], out: [O][HW]
// ---------------------------------------------------------------------------
template<int C, int O>
__global__ __launch_bounds__(256)
void dcn3d_k(const float* __restrict__ frames, const float* __restrict__ raw,
             const float* __restrict__ w, const float* __restrict__ bias,
             float* __restrict__ out)
{
    int p = blockIdx.x * blockDim.x + threadIdx.x;
    if (p >= HW) return;
    const float yf = (float)(p >> 8);
    const float xf = (float)(p & 255);

    float acc[O];
#pragma unroll
    for (int o = 0; o < O; ++o) acc[o] = bias[o];

    for (int n = 0; n < 100; ++n) {
        int d  = n / 25;
        int t  = n - d * 25;
        int ky = t / 5;
        int kx = t - ky * 5;

        float od = raw[n * HW + p];
        float oy = raw[(100 + n) * HW + p];
        float ox = raw[(200 + n) * HW + p];
        float mk = sigmoidf_(raw[(300 + n) * HW + p]);

        float pd = fminf(fmaxf((float)d + od, 0.0f), 3.0f);
        float py = fminf(fmaxf((float)(ky - 2) + yf + oy, 0.0f), 255.0f);
        float px = fminf(fmaxf((float)(kx - 2) + xf + ox, 0.0f), 255.0f);

        float fd0 = floorf(pd), fy0 = floorf(py), fx0 = floorf(px);
        int d0 = (int)fd0, y0 = (int)fy0, x0 = (int)fx0;
        float fd = pd - fd0, fy = py - fy0, fx = px - fx0;
        int d1 = min(d0 + 1, 3), y1 = min(y0 + 1, 255), x1 = min(x0 + 1, 255);

        float w00 = (1.f - fy) * (1.f - fx), w01 = (1.f - fy) * fx;
        float w10 = fy * (1.f - fx),         w11 = fy * fx;
        int i00 = y0 * WDIM + x0, i01 = y0 * WDIM + x1;
        int i10 = y1 * WDIM + x0, i11 = y1 * WDIM + x1;

        const float* fr0 = frames + d0 * (C * HW);
        const float* fr1 = frames + d1 * (C * HW);

        float val[C];
#pragma unroll
        for (int c = 0; c < C; ++c) {
            int cb = c * HW;
            float v0 = fr0[cb + i00] * w00 + fr0[cb + i01] * w01
                     + fr0[cb + i10] * w10 + fr0[cb + i11] * w11;
            float v1 = fr1[cb + i00] * w00 + fr1[cb + i01] * w01
                     + fr1[cb + i10] * w10 + fr1[cb + i11] * w11;
            val[c] = (v0 * (1.f - fd) + v1 * fd) * mk;
        }
#pragma unroll
        for (int o = 0; o < O; ++o)
#pragma unroll
            for (int c = 0; c < C; ++c)
                acc[o] += val[c] * w[(o * C + c) * 100 + n];   // uniform -> scalar loads
    }
#pragma unroll
    for (int o = 0; o < O; ++o) out[o * HW + p] = acc[o];
}

// ---------------------------------------------------------------------------
// Coordinate attention. pool/y/attn: [C or MID][512]; pos<256 = H, else W.
// ---------------------------------------------------------------------------
__global__ void ca_pool_k(const float* __restrict__ x, float* __restrict__ pool, int C)
{
    int tid = blockIdx.x * blockDim.x + threadIdx.x;
    if (tid >= C * 512) return;
    int c = tid >> 9, pos = tid & 511;
    float s = 0.0f;
    if (pos < HDIM) {
        const float* row = x + c * HW + pos * WDIM;
        for (int i = 0; i < WDIM; ++i) s += row[i];
    } else {
        const float* col = x + c * HW + (pos - HDIM);
        for (int i = 0; i < HDIM; ++i) s += col[i * WDIM];
    }
    pool[tid] = s * (1.0f / 256.0f);
}

__global__ void ca_mid_k(const float* __restrict__ pool, const float* __restrict__ w1,
                         const float* __restrict__ b1, float* __restrict__ y,
                         int C, int MID)
{
    int tid = blockIdx.x * blockDim.x + threadIdx.x;
    if (tid >= MID * 512) return;
    int m = tid >> 9, pos = tid & 511;
    float s = b1[m];
    for (int c = 0; c < C; ++c) s += w1[m * C + c] * pool[(c << 9) + pos];
    y[tid] = hswishf_(s);
}

__global__ void ca_attn_k(const float* __restrict__ y,
                          const float* __restrict__ wh, const float* __restrict__ bh,
                          const float* __restrict__ ww, const float* __restrict__ bw,
                          float* __restrict__ attn, int C, int MID)
{
    int tid = blockIdx.x * blockDim.x + threadIdx.x;
    if (tid >= C * 512) return;
    int c = tid >> 9, pos = tid & 511;
    const float* wsel = (pos < HDIM) ? wh : ww;
    float s = (pos < HDIM) ? bh[c] : bw[c];
    for (int m = 0; m < MID; ++m) s += wsel[c * MID + m] * y[(m << 9) + pos];
    attn[tid] = sigmoidf_(s);
}

__global__ void ca_apply_k(float* __restrict__ x, const float* __restrict__ attn, int C)
{
    int tid = blockIdx.x * blockDim.x + threadIdx.x;
    if (tid >= C * HW) return;
    int c = tid >> 16, p = tid & (HW - 1);
    int yy = p >> 8, xx = p & 255;
    x[tid] *= attn[(c << 9) + yy] * attn[(c << 9) + HDIM + xx];
}

static void coord_att(float* x, int C, int MID,
                      const float* w1, const float* b1,
                      const float* wh, const float* bh,
                      const float* ww, const float* bw,
                      float* pool, float* ymid, float* attn, hipStream_t s)
{
    ca_pool_k<<<(C * 512 + 255) / 256, 256, 0, s>>>(x, pool, C);
    ca_mid_k<<<(MID * 512 + 255) / 256, 256, 0, s>>>(pool, w1, b1, ymid, C, MID);
    ca_attn_k<<<(C * 512 + 255) / 256, 256, 0, s>>>(ymid, wh, bh, ww, bw, attn, C, MID);
    ca_apply_k<<<(C * HW + 255) / 256, 256, 0, s>>>(x, attn, C);
}

// ---------------------------------------------------------------------------
__global__ void pack4_k(const float* __restrict__ a, const float* __restrict__ b,
                        const float* __restrict__ c, const float* __restrict__ d,
                        float* __restrict__ out)
{
    int i = blockIdx.x * blockDim.x + threadIdx.x;
    if (i >= 3 * HW) return;
    out[i]          = a[i];
    out[i + 3 * HW] = b[i];
    out[i + 6 * HW] = c[i];
    out[i + 9 * HW] = d[i];
}

__global__ void copy_k(const float* __restrict__ src, float* __restrict__ dst, int n)
{
    int i = blockIdx.x * blockDim.x + threadIdx.x;
    if (i < n) dst[i] = src[i];
}

// ---------------------------------------------------------------------------
// Input index map (setup_inputs() flattening order).
enum {
    I_IMG1 = 0, I_IMG2, I_IMG4, I_IMG5,
    I_OFF_W1, I_OFF_B1, I_OFF_W2, I_OFF_B2, I_OFF_W3, I_OFF_B3,
    I_DCN_IMG_W, I_DCN_IMG_B, I_DCN_CTX_W, I_DCN_CTX_B,
    I_CN_W1, I_CN_B1, I_CN_W2, I_CN_B2, I_CN_W3, I_CN_B3,
    I_CAC_W1, I_CAC_B1, I_CAC_WH, I_CAC_BH, I_CAC_WW, I_CAC_BW,
    I_CTXO_W, I_CTXO_B,
    I_CAS_W1, I_CAS_B1, I_CAS_WH, I_CAS_BH, I_CAS_WW, I_CAS_BW,
    I_GIN_W, I_GIN_B, I_R1A_W, I_R1A_B, I_R1B_W, I_R1B_B,
    I_R2A_W, I_R2A_B, I_R2B_W, I_R2B_B, I_GOUT_W, I_GOUT_B
};

extern "C" void kernel_launch(void* const* d_in, const int* in_sizes, int n_in,
                              void* d_out, int out_size, void* d_ws, size_t ws_size,
                              hipStream_t stream)
{
    (void)in_sizes; (void)n_in; (void)out_size; (void)ws_size;
#define P(i) ((const float*)d_in[(i)])

    float* ws = (float*)d_ws;
    // Workspace layout (floats), ~154 MB — L2/HBM resident throughout.
    float* cat4 = ws;                      // [12][HW]  (also the 4 DCN img frames)
    float* h1   = cat4 + 12 * HW;          // [64][HW]  (reused as ctx48 [48][HW])
    float* h2   = h1 + 64 * HW;            // [64][HW]  (reused as ctxbuf [4][6][HW])
    float* raw  = h2 + 64 * HW;            // [400][HW]
    float* mid  = raw + 400 * HW;          // [3][HW]
    float* se   = mid + 3 * HW;            // [9][HW]
    float* g    = se + 9 * HW;             // [16][HW]
    float* r    = g + 16 * HW;             // [16][HW]
    float* pool = r + 16 * HW;             // [48][512]
    float* ymid = pool + 48 * 512;         // [16][512]
    float* attn = ymid + 16 * 512;         // [48][512]
    _Float16* bpack = (_Float16*)(attn + 48 * 512);  // max 25*18*512 f16 = 450 KB

    float* ctx48  = h1;
    float* ctxbuf = h2;
    float* out    = (float*)d_out;

    // ---- gather frames: cat4 = [img1|img2|img4|img5] along channels ----
    pack4_k<<<(3 * HW + 255) / 256, 256, 0, stream>>>(
        P(I_IMG1), P(I_IMG2), P(I_IMG4), P(I_IMG5), cat4);

    // ---- offset head: 12->64 relu, 64->64 relu, 64->400 ----
    conv3x3<32, 12, 1, false>(cat4, P(I_OFF_W1), P(I_OFF_B1), nullptr, h1, 64, bpack, stream);
    conv3x3<64, 64, 1, false>(h1,   P(I_OFF_W2), P(I_OFF_B2), nullptr, h2, 64, bpack, stream);
    conv3x3<64, 64, 0, false>(h2,   P(I_OFF_W3), P(I_OFF_B3), nullptr, raw, 400, bpack, stream);

    // ---- image DCN3D -> mid [3][HW] ----
    dcn3d_k<3, 3><<<(HW + 255) / 256, 256, 0, stream>>>(
        cat4, raw, P(I_DCN_IMG_W), P(I_DCN_IMG_B), mid);

    // ---- per-frame context nets + coord-att + 48->6 projection ----
    for (int i = 0; i < 4; ++i) {
        const float* img = cat4 + i * 3 * HW;
        conv3x3<32, 3, 1, false>(img,             P(I_CN_W1), P(I_CN_B1), nullptr, ctx48,           16, bpack, stream);
        conv3x3<32, 16, 1, false>(ctx48,           P(I_CN_W2), P(I_CN_B2), nullptr, ctx48 + 16 * HW, 16, bpack, stream);
        conv3x3<32, 16, 1, false>(ctx48 + 16 * HW, P(I_CN_W3), P(I_CN_B3), nullptr, ctx48 + 32 * HW, 16, bpack, stream);
        coord_att(ctx48, 48, 16,
                  P(I_CAC_W1), P(I_CAC_B1), P(I_CAC_WH), P(I_CAC_BH),
                  P(I_CAC_WW), P(I_CAC_BW), pool, ymid, attn, stream);
        conv3x3<64, 48, 0, false>(ctx48, P(I_CTXO_W), P(I_CTXO_B), nullptr,
                                  ctxbuf + i * 6 * HW, 6, bpack, stream);
    }

    // ---- context DCN3D -> se ch 3..8 ; mid -> se ch 0..2 ----
    dcn3d_k<6, 6><<<(HW + 255) / 256, 256, 0, stream>>>(
        ctxbuf, raw, P(I_DCN_CTX_W), P(I_DCN_CTX_B), se + 3 * HW);
    copy_k<<<(3 * HW + 255) / 256, 256, 0, stream>>>(mid, se, 3 * HW);

    // ---- coord-att on se (C=9, mid=8) ----
    coord_att(se, 9, 8,
              P(I_CAS_W1), P(I_CAS_B1), P(I_CAS_WH), P(I_CAS_BH),
              P(I_CAS_WW), P(I_CAS_BW), pool, ymid, attn, stream);

    // ---- grid refinement net ----
    conv3x3<32, 9, 1, false>(se, P(I_GIN_W), P(I_GIN_B), nullptr, g, 16, bpack, stream);
    conv3x3<32, 16, 1, false>(g, P(I_R1A_W), P(I_R1A_B), nullptr, r, 16, bpack, stream);
    conv3x3<32, 16, 0, true >(r, P(I_R1B_W), P(I_R1B_B), g, g, 16, bpack, stream);   // g += conv(r)
    conv3x3<32, 16, 1, false>(g, P(I_R2A_W), P(I_R2A_B), nullptr, r, 16, bpack, stream);
    conv3x3<32, 16, 0, true >(r, P(I_R2B_W), P(I_R2B_B), g, g, 16, bpack, stream);   // g += conv(r)

    // ---- final: out = mid + conv(g, g_out) ----
    conv3x3<32, 16, 0, true>(g, P(I_GOUT_W), P(I_GOUT_B), mid, out, 3, bpack, stream);
#undef P
}
